// Attention_15023795602055
// MI455X (gfx1250) — compile-verified, matrix operand order checked by reading, not run on hardware
//
#include <hip/hip_runtime.h>

#define DIM  2048
#define NH   16
#define HD   128
#define BB   2
#define SS   2048
#define E3   (3 * DIM)

typedef _Float16 half8 __attribute__((ext_vector_type(8)));
typedef _Float16 v16h  __attribute__((ext_vector_type(16)));
typedef float    v8f   __attribute__((ext_vector_type(8)));

// ---------- helpers ----------

__device__ __forceinline__ v8f vzero() {
    v8f z;
#pragma unroll
    for (int i = 0; i < 8; ++i) z[i] = 0.0f;
    return z;
}

// Assemble a 16-half A/B fragment from two contiguous 8-half (16B) chunks.
// CDNA5 16-bit fragment layout: lane holds K = koff+0..7 and koff+16..23.
__device__ __forceinline__ v16h make_frag(const _Float16* p0, const _Float16* p1) {
    half8 lo = *(const half8*)p0;
    half8 hi = *(const half8*)p1;
    v16h r;
#pragma unroll
    for (int i = 0; i < 8; ++i) { r[i] = lo[i]; r[i + 8] = hi[i]; }
    return r;
}

__device__ __forceinline__ v8f wmma_f16(v16h a, v16h b, v8f c) {
    // (neg_a, A, neg_b, B, c_mod, C, reuse_a, reuse_b)
    return __builtin_amdgcn_wmma_f32_16x16x32_f16(false, a, false, b, (short)0, c,
                                                  false, false);
}

__device__ __forceinline__ half8 cvt8(float4 a, float4 b) {
    half8 h;
    h[0] = (_Float16)a.x; h[1] = (_Float16)a.y; h[2] = (_Float16)a.z; h[3] = (_Float16)a.w;
    h[4] = (_Float16)b.x; h[5] = (_Float16)b.y; h[6] = (_Float16)b.z; h[7] = (_Float16)b.w;
    return h;
}

// ---------- async global->LDS (CDNA5 GLOBAL_LOAD_ASYNC_TO_LDS_B128) ----------
// The clang builtin wants CUDA language-AS-qualified pointers (unspellable in
// HIP), so emit the instruction directly. LDS operand = low 32 bits of the
// generic shared pointer (ISA 10.2: generic->LDS truncates to addr[31:0]);
// global operand = 64-bit VGPR pair; "off" = no-saddr (GV) form.

#if defined(__gfx1250__)
#define ASYNC_LDS 1
#endif

__device__ __forceinline__ void async_cp16(const _Float16* g, _Float16* l) {
#ifdef ASYNC_LDS
    const uint32_t lofs = (uint32_t)(uintptr_t)l;
    asm volatile("global_load_async_to_lds_b128 %0, %1, off"
                 :: "v"(lofs), "v"(g) : "memory");
#else
    *(half8*)l = *(const half8*)g;
#endif
}

__device__ __forceinline__ void async_wait() {
#ifdef ASYNC_LDS
    asm volatile("s_wait_asynccnt 0x0" ::: "memory");
#endif
}

// ---------- kernel 1: qkv = x @ wqkv^T, RoPE on q/k, scatter to Qh/Kh/Vt ----------
// Qh, Kh: [B, NH, S, HD] f16 ;  Vt: [B, NH, HD, S] f16 (transposed for P@V B-frags)

__global__ __launch_bounds__(256) void qkv_rope_kernel(
    const float* __restrict__ x, const float* __restrict__ freqs,
    const float* __restrict__ wqkv,
    _Float16* __restrict__ qh, _Float16* __restrict__ kh, _Float16* __restrict__ vt)
{
    __shared__ _Float16 ldsA[2][128 * 32];
    __shared__ _Float16 ldsB[2][128 * 32];

    const int tid  = threadIdx.x;
    const int lane = tid & 31;
    const int wid  = tid >> 5;
    const int wm   = wid & 3;    // 0..3 -> 32 rows each
    const int wn   = wid >> 2;   // 0..1 -> 64 cols each
    const int tileM = blockIdx.y * 128;
    const int tileN = blockIdx.x * 128;

    const int ln   = lane & 15;
    const int kofs = (lane >> 4) * 8;
    const int srow = tid >> 1;          // staging row 0..127
    const int scol = (tid & 1) * 16;    // staging col 0 / 16

    const float* aBase = x    + (size_t)(tileM + srow) * DIM + scol;
    const float* bBase = wqkv + (size_t)(tileN + srow) * DIM + scol;

    v8f acc[2][4];
#pragma unroll
    for (int i = 0; i < 2; ++i)
#pragma unroll
        for (int j = 0; j < 4; ++j) acc[i][j] = vzero();

    float4 ra0, ra1, ra2, ra3, rb0, rb1, rb2, rb3;
    auto load_regs = [&](int kt) {
        const float* ap = aBase + kt;
        const float* bp = bBase + kt;
        ra0 = *(const float4*)(ap + 0); ra1 = *(const float4*)(ap + 4);
        ra2 = *(const float4*)(ap + 8); ra3 = *(const float4*)(ap + 12);
        rb0 = *(const float4*)(bp + 0); rb1 = *(const float4*)(bp + 4);
        rb2 = *(const float4*)(bp + 8); rb3 = *(const float4*)(bp + 12);
    };
    auto store_lds = [&](int p) {
        *(half8*)&ldsA[p][srow * 32 + scol + 0] = cvt8(ra0, ra1);
        *(half8*)&ldsA[p][srow * 32 + scol + 8] = cvt8(ra2, ra3);
        *(half8*)&ldsB[p][srow * 32 + scol + 0] = cvt8(rb0, rb1);
        *(half8*)&ldsB[p][srow * 32 + scol + 8] = cvt8(rb2, rb3);
    };

    // software pipeline: prologue
    load_regs(0);
    store_lds(0);
    __syncthreads();

    int p = 0;
    for (int kt = 0; kt < DIM; kt += 32) {
        const bool more = (kt + 32) < DIM;
        if (more) {
            load_regs(kt + 32);                       // overlaps WMMAs below
            if (kt + 64 < DIM) {
                __builtin_prefetch(aBase + kt + 64, 0, 1);
                __builtin_prefetch(bBase + kt + 64, 0, 1);
            }
        }

        v16h af[2], bf[4];
#pragma unroll
        for (int mt = 0; mt < 2; ++mt) {
            const int ar = wm * 32 + mt * 16 + ln;
            af[mt] = make_frag(&ldsA[p][ar * 32 + kofs], &ldsA[p][ar * 32 + kofs + 16]);
        }
#pragma unroll
        for (int nt = 0; nt < 4; ++nt) {
            const int br = wn * 64 + nt * 16 + ln;
            bf[nt] = make_frag(&ldsB[p][br * 32 + kofs], &ldsB[p][br * 32 + kofs + 16]);
        }
#pragma unroll
        for (int mt = 0; mt < 2; ++mt)
#pragma unroll
            for (int nt = 0; nt < 4; ++nt)
                acc[mt][nt] = wmma_f16(af[mt], bf[nt], acc[mt][nt]);

        if (more) store_lds(p ^ 1);
        __syncthreads();
        p ^= 1;
    }

    // epilogue: RoPE (q,k) via cross-lane pair exchange, scatter to ws
#pragma unroll
    for (int mt = 0; mt < 2; ++mt) {
#pragma unroll
        for (int nt = 0; nt < 4; ++nt) {
            const int ecol = tileN + wn * 64 + nt * 16 + ln;   // 0..6143
            const int part = ecol >> 11;                       // 0=q 1=k 2=v (block-uniform)
            const int c    = ecol & (DIM - 1);
            const int h    = c >> 7;
            const int hd   = c & (HD - 1);
#pragma unroll
            for (int r = 0; r < 8; ++r) {
                const int m = tileM + wm * 32 + mt * 16 + r + ((lane >> 4) << 3);
                const int b = m >> 11;
                const int s = m & (SS - 1);
                float v = acc[mt][nt][r];
                if (part < 2) {
                    const int j = hd >> 1;
                    const float2 fc = *(const float2*)(freqs + ((size_t)s * (HD / 2) + j) * 2);
                    const float other = __shfl_xor(v, 1, 32);
                    v = (hd & 1) ? (v * fc.x + other * fc.y)
                                 : (v * fc.x - other * fc.y);
                    _Float16* dst = (part == 0) ? qh : kh;
                    dst[(((size_t)b * NH + h) * SS + s) * HD + hd] = (_Float16)v;
                } else {
                    vt[(((size_t)b * NH + h) * HD + hd) * SS + s] = (_Float16)v;
                }
            }
        }
    }
}

// ---------- kernel 2: flash attention per (b,h); 4 waves x 16 Q rows ----------
// ao: [B, S, DIM] f16 (attention output, GEMM-ready A layout for out proj)

__global__ __launch_bounds__(128) void attn_kernel(
    const _Float16* __restrict__ qh, const _Float16* __restrict__ kh,
    const _Float16* __restrict__ vt, _Float16* __restrict__ ao)
{
    __shared__ _Float16 ldsK[2][64 * 128];   // [s'][hd], double buffered
    __shared__ _Float16 ldsV[2][128 * 64];   // [hd][s'] (transposed V)
    __shared__ _Float16 ldsP[4][16 * 64];    // per-wave P staging

    const int tid   = threadIdx.x;
    const int lane  = tid & 31;
    const int w     = tid >> 5;             // 0..3
    const int qbase = blockIdx.x * 64;
    const int bh    = blockIdx.y;
    const int b     = bh >> 4;
    const int h     = bh & 15;

    const _Float16* qp = qh + (size_t)bh * SS * HD;
    const _Float16* kp = kh + (size_t)bh * SS * HD;
    const _Float16* vp = vt + (size_t)bh * HD * SS;

    const int ln   = lane & 15;
    const int kofs = (lane >> 4) * 8;

    // issue async copies of K block kb into buffer pb
    auto issue_block = [&](int kb, int pb) {
        const _Float16* kblk = kp + (size_t)kb * 64 * HD;   // contiguous 16KB
#pragma unroll
        for (int i = 0; i < 8; ++i)
            async_cp16(kblk + (i * 128 + tid) * 8, &ldsK[pb][(i * 128 + tid) * 8]);
        const _Float16* vrow = vp + (size_t)tid * SS + kb * 64;
#pragma unroll
        for (int i = 0; i < 8; ++i)
            async_cp16(vrow + i * 8, &ldsV[pb][tid * 64 + i * 8]);
    };

    // private Q fragments: 16 rows x 128 (4 k-chunks)
    v16h qf[4];
    {
        const int qr = qbase + w * 16 + ln;
#pragma unroll
        for (int kc = 0; kc < 4; ++kc) {
            const _Float16* ptr = qp + (size_t)qr * HD + kc * 32 + kofs;
            qf[kc] = make_frag(ptr, ptr + 16);
        }
    }

    float mrow[8], lrow[8];
    v8f oacc[8];
#pragma unroll
    for (int r = 0; r < 8; ++r) { mrow[r] = -1e30f; lrow[r] = 0.0f; }
#pragma unroll
    for (int ot = 0; ot < 8; ++ot) oacc[ot] = vzero();

    const float scale = 0.08838834764831845f;   // 1/sqrt(128)

    issue_block(0, 0);

    for (int kb = 0; kb < SS / 64; ++kb) {
        const int p = kb & 1;
        async_wait();               // drain this wave's async copies into buf p
        __syncthreads();            // buf p visible to all waves; buf p^1 free
        if (kb + 1 < SS / 64)
            issue_block(kb + 1, p ^ 1);   // DMA next block during compute

        // scores: 16 x 64 = 4 n-tiles x 4 k-chunks
        v8f st[4];
#pragma unroll
        for (int nt = 0; nt < 4; ++nt) {
            v8f a = vzero();
#pragma unroll
            for (int kc = 0; kc < 4; ++kc) {
                const int kr = nt * 16 + ln;    // B-frag column = K-row s'
                v16h bf = make_frag(&ldsK[p][kr * 128 + kc * 32 + kofs],
                                    &ldsK[p][kr * 128 + kc * 32 + kofs + 16]);
                a = wmma_f16(qf[kc], bf, a);
            }
            st[nt] = a;
        }

        // online softmax; row (r, lane>>4) groups reduce over 16 lanes
#pragma unroll
        for (int r = 0; r < 8; ++r) {
            float mx = -1e30f;
#pragma unroll
            for (int nt = 0; nt < 4; ++nt) { st[nt][r] *= scale; mx = fmaxf(mx, st[nt][r]); }
#pragma unroll
            for (int off = 1; off < 16; off <<= 1)
                mx = fmaxf(mx, __shfl_xor(mx, off, 32));
            const float mn    = fmaxf(mrow[r], mx);
            const float alpha = __expf(mrow[r] - mn);
            mrow[r] = mn;
            float ps = 0.0f;
#pragma unroll
            for (int nt = 0; nt < 4; ++nt) {
                const float pe = __expf(st[nt][r] - mn);
                st[nt][r] = pe;
                ps += pe;
            }
#pragma unroll
            for (int off = 1; off < 16; off <<= 1)
                ps += __shfl_xor(ps, off, 32);
            lrow[r] = lrow[r] * alpha + ps;
#pragma unroll
            for (int ot = 0; ot < 8; ++ot) oacc[ot][r] *= alpha;
        }

        // P (C-layout) -> per-wave LDS tile [16][64] f16
#pragma unroll
        for (int nt = 0; nt < 4; ++nt)
#pragma unroll
            for (int r = 0; r < 8; ++r)
                ldsP[w][(r + ((lane >> 4) << 3)) * 64 + nt * 16 + ln] =
                    (_Float16)st[nt][r];
        __syncthreads();

        // P @ V: 8 out-tiles x 2 k-chunks
#pragma unroll
        for (int kc = 0; kc < 2; ++kc) {
            v16h pf = make_frag(&ldsP[w][ln * 64 + kc * 32 + kofs],
                                &ldsP[w][ln * 64 + kc * 32 + kofs + 16]);
#pragma unroll
            for (int ot = 0; ot < 8; ++ot) {
                const int vr = ot * 16 + ln;    // B-frag column = hd
                v16h vf = make_frag(&ldsV[p][vr * 64 + kc * 32 + kofs],
                                    &ldsV[p][vr * 64 + kc * 32 + kofs + 16]);
                oacc[ot] = wmma_f16(pf, vf, oacc[ot]);
            }
        }
    }

    // normalize and store to ao[b][s][h*128+hd]
#pragma unroll
    for (int r = 0; r < 8; ++r) {
        const float inv = 1.0f / lrow[r];
        const int s = qbase + w * 16 + r + ((lane >> 4) << 3);
#pragma unroll
        for (int ot = 0; ot < 8; ++ot) {
            const int hd = ot * 16 + ln;
            ao[((size_t)b * SS + s) * DIM + h * HD + hd] = (_Float16)(oacc[ot][r] * inv);
        }
    }
}

// ---------- kernel 3: out = ao @ wo^T (f32 output) ----------

__global__ __launch_bounds__(256) void oproj_kernel(
    const _Float16* __restrict__ ao, const float* __restrict__ wo,
    float* __restrict__ out)
{
    __shared__ _Float16 ldsA[2][128 * 32];
    __shared__ _Float16 ldsB[2][128 * 32];

    const int tid  = threadIdx.x;
    const int lane = tid & 31;
    const int wid  = tid >> 5;
    const int wm   = wid & 3;
    const int wn   = wid >> 2;
    const int tileM = blockIdx.y * 128;
    const int tileN = blockIdx.x * 128;

    const int ln   = lane & 15;
    const int kofs = (lane >> 4) * 8;
    const int srow = tid >> 1;
    const int scol = (tid & 1) * 16;

    const _Float16* aBase = ao + (size_t)(tileM + srow) * DIM + scol;
    const float*    bBase = wo + (size_t)(tileN + srow) * DIM + scol;

    v8f acc[2][4];
#pragma unroll
    for (int i = 0; i < 2; ++i)
#pragma unroll
        for (int j = 0; j < 4; ++j) acc[i][j] = vzero();

    half8 ha0, ha1;
    float4 rb0, rb1, rb2, rb3;
    auto load_regs = [&](int kt) {
        ha0 = *(const half8*)(aBase + kt + 0);
        ha1 = *(const half8*)(aBase + kt + 8);
        const float* bp = bBase + kt;
        rb0 = *(const float4*)(bp + 0); rb1 = *(const float4*)(bp + 4);
        rb2 = *(const float4*)(bp + 8); rb3 = *(const float4*)(bp + 12);
    };
    auto store_lds = [&](int pp) {
        *(half8*)&ldsA[pp][srow * 32 + scol + 0] = ha0;
        *(half8*)&ldsA[pp][srow * 32 + scol + 8] = ha1;
        *(half8*)&ldsB[pp][srow * 32 + scol + 0] = cvt8(rb0, rb1);
        *(half8*)&ldsB[pp][srow * 32 + scol + 8] = cvt8(rb2, rb3);
    };

    load_regs(0);
    store_lds(0);
    __syncthreads();

    int p = 0;
    for (int kt = 0; kt < DIM; kt += 32) {
        const bool more = (kt + 32) < DIM;
        if (more) {
            load_regs(kt + 32);
            if (kt + 64 < DIM) {
                __builtin_prefetch(aBase + kt + 64, 0, 1);
                __builtin_prefetch(bBase + kt + 64, 0, 1);
            }
        }

        v16h af[2], bf[4];
#pragma unroll
        for (int mt = 0; mt < 2; ++mt) {
            const int ar = wm * 32 + mt * 16 + ln;
            af[mt] = make_frag(&ldsA[p][ar * 32 + kofs], &ldsA[p][ar * 32 + kofs + 16]);
        }
#pragma unroll
        for (int nt = 0; nt < 4; ++nt) {
            const int br = wn * 64 + nt * 16 + ln;
            bf[nt] = make_frag(&ldsB[p][br * 32 + kofs], &ldsB[p][br * 32 + kofs + 16]);
        }
#pragma unroll
        for (int mt = 0; mt < 2; ++mt)
#pragma unroll
            for (int nt = 0; nt < 4; ++nt)
                acc[mt][nt] = wmma_f16(af[mt], bf[nt], acc[mt][nt]);

        if (more) store_lds(p ^ 1);
        __syncthreads();
        p ^= 1;
    }

#pragma unroll
    for (int mt = 0; mt < 2; ++mt)
#pragma unroll
        for (int nt = 0; nt < 4; ++nt) {
            const int ecol = tileN + wn * 64 + nt * 16 + ln;
#pragma unroll
            for (int r = 0; r < 8; ++r) {
                const int m = tileM + wm * 32 + mt * 16 + r + ((lane >> 4) << 3);
                out[(size_t)m * DIM + ecol] = acc[mt][nt][r];
            }
        }
}

// ---------- launch ----------

extern "C" void kernel_launch(void* const* d_in, const int* in_sizes, int n_in,
                              void* d_out, int out_size, void* d_ws, size_t ws_size,
                              hipStream_t stream) {
    (void)in_sizes; (void)n_in; (void)out_size; (void)ws_size;
    const float* x     = (const float*)d_in[0];
    const float* freqs = (const float*)d_in[1];
    const float* wqkv  = (const float*)d_in[2];
    const float* wo    = (const float*)d_in[3];
    float* out = (float*)d_out;

    const size_t QK = (size_t)BB * NH * SS * HD;   // 8.4M halfs per tensor
    _Float16* qh = (_Float16*)d_ws;
    _Float16* kh = qh + QK;
    _Float16* vt = kh + QK;
    _Float16* ao = vt + QK;                        // [B, S, DIM] f16

    dim3 b1(256), g1(E3 / 128, (BB * SS) / 128);   // 48 x 32
    qkv_rope_kernel<<<g1, b1, 0, stream>>>(x, freqs, wqkv, qh, kh, vt);

    dim3 b2(128), g2(SS / 64, BB * NH);            // 32 x 32
    attn_kernel<<<g2, b2, 0, stream>>>(qh, kh, vt, ao);

    dim3 b3(256), g3(DIM / 128, (BB * SS) / 128);  // 16 x 32
    oproj_kernel<<<g3, b3, 0, stream>>>(ao, wo, out);
}
